// PushTModel_11622181503025
// MI455X (gfx1250) — matrix operand order repaired and checked by reading, not compiled, not run on hardware
//
#include <hip/hip_runtime.h>

typedef __attribute__((ext_vector_type(16))) _Float16 v16h;
typedef __attribute__((ext_vector_type(8)))  float    v8f;

#define NBODY   2
#define MDIM    8
#define EDIM    192
#define TSEQ    5
#define NSTEP   4
#define BROWS   32768
#define NTILES  (BROWS/16)

#define DT_C        (5.0f/60.0f)
#define CONTACT_C   0.05f
#define INNOV_C     0.3f
#define TAU_C       0.01f
#define EPS_C       1e-6f

// packed f16 weight-fragment sections (offsets in halves from wsH base)
#define OFF_ZW1 0
#define OFF_ZW2 24576
#define OFF_SW1 26624
#define OFF_SW2 51200
#define OFF_FW1 53248
#define OFF_FW2 55296
#define OFF_FW3 59392
#define OFF_OW1 60416
#define OFF_OW2 68608

// LDS strides (halves); all multiples of 8 so 16-byte b128 loads stay aligned
#define AS 200
#define GS 264
#define FS 40

// ---- branch-free fast transcendentals (v_exp_f32 / v_log_f32 paths) ----
__device__ __forceinline__ float exp_fast(float x){ return __expf(x); }
__device__ __forceinline__ float tanh_fast(float u){
  // 1 - 2/(1+e^{2u}); exp saturates to +inf / 0 -> tanh saturates to +/-1. No branches.
  return 1.0f - 2.0f/(1.0f + exp_fast(2.0f*u));
}
__device__ __forceinline__ float gelu_t(float x){
  const float c0 = 0.7978845608028654f;               // sqrt(2/pi), jax approximate=True
  return 0.5f*x*(1.0f + tanh_fast(c0*(x + 0.044715f*x*x*x)));
}
__device__ __forceinline__ float softplus_f(float x){
  return fmaxf(x, 0.0f) + __logf(1.0f + exp_fast(-fabsf(x)));
}
__device__ __forceinline__ float sigmoid_f(float x){
  return 1.0f/(1.0f + exp_fast(-x));
}
__device__ __forceinline__ v8f splat8(float b){
  v8f r;
#pragma unroll
  for (int i=0;i<8;++i) r[i]=b;
  return r;
}
__device__ __forceinline__ v8f wmma_f16(v16h a, v16h b, v8f c){
  // D = A(16x32,f16) x B(32x16,f16) + C(16x16,f32)
  return __builtin_amdgcn_wmma_f32_16x16x32_f16(false, a, false, b, (short)0, c, false, false);
}

// A fragment (ISA 16-bit A 16x32 layout): lane L -> row m=L%16,
// halves 0..7 : K = k0 + (L/16)*8 + 0..7 ; halves 8..15 : +16.
__device__ __forceinline__ v16h frag_a(const _Float16* base, int m, int stride,
                                       int cbase, int k0, int lane){
  const _Float16* p = base + m*stride + cbase + k0 + ((lane>>4)<<3);
  union { v16h v; uint4 u[2]; } r;
  r.u[0] = *reinterpret_cast<const uint4*>(p);
  r.u[1] = *reinterpret_cast<const uint4*>(p + 16);
  return r.v;
}
// B fragment from pre-swizzled global section: 16 contiguous halves per lane.
__device__ __forceinline__ v16h frag_b(const _Float16* w, int kt, int nt, int NT, int lane){
  const _Float16* p = w + (size_t)((((kt*NT)+nt)<<5) + lane)*16;
  union { v16h v; uint4 u[2]; } r;
  r.u[0] = *reinterpret_cast<const uint4*>(p);
  r.u[1] = *reinterpret_cast<const uint4*>(p + 8);
  return r.v;
}

// ---------- weight pack: fp32 (K,N) row-major -> f16 WMMA B-fragments ----------
// B layout (matches frag_a): K = kt*32 + (lane/16)*8 + (h&7) + (h>=8 ? 16 : 0)
// (h=0..15) ; N = nt*16 + lane%16. Zero-pad OOB.
__global__ void pack_wfrag(const float* __restrict__ src, _Float16* __restrict__ dst,
                           int Kreal, int Nreal, int NT, int total){
  int idx = blockIdx.x*blockDim.x + threadIdx.x;
  if (idx >= total) return;
  int h    =  idx        & 15;
  int lane = (idx >> 4)  & 31;
  int tile =  idx >> 9;
  int nt   = tile % NT;
  int kt   = tile / NT;
  int n = nt*16 + (lane & 15);
  int k = kt*32 + ((lane >> 4) << 3) + (h & 7) + ((h >> 3) << 4);
  float v = (k < Kreal && n < Nreal) ? src[(size_t)k*Nreal + n] : 0.0f;
  dst[idx] = (_Float16)v;
}

// ---------- 192->128(gelu)->16 encoder on a 16-row tile ----------
__device__ __forceinline__ void encoder16(
    const _Float16* aS, _Float16* gS, float* cS,
    const _Float16* w1, const _Float16* w2,
    const float* __restrict__ b1, const float* __restrict__ b2,
    int lane, bool do_sp, float out[8])
{
  int lm = lane & 15;
  int hb = (lane >> 4) << 3;               // C-tile row base for this lane half
#pragma unroll
  for (int nt=0; nt<8; ++nt){
    v8f acc = splat8(b1[nt*16 + lm]);
#pragma unroll
    for (int kt=0; kt<6; ++kt)
      acc = wmma_f16(frag_a(aS, lm, AS, 0, kt*32, lane), frag_b(w1, kt, nt, 8, lane), acc);
    int cc = nt*16 + lm;
#pragma unroll
    for (int r=0;r<8;++r) gS[(hb+r)*GS + cc] = (_Float16)gelu_t(acc[r]);
  }
  __syncthreads();
  {
    v8f acc = splat8(b2[lm]);
#pragma unroll
    for (int kt=0; kt<4; ++kt)
      acc = wmma_f16(frag_a(gS, lm, GS, 0, kt*32, lane), frag_b(w2, kt, 0, 1, lane), acc);
#pragma unroll
    for (int r=0;r<8;++r) cS[(hb+r)*16 + lm] = acc[r];
  }
  __syncthreads();
#pragma unroll
  for (int i=0;i<8;++i){
    float v = cS[lm*16 + hb + i];          // col = body*8 + i, row = lm
    out[i] = do_sp ? softplus_f(v) : v;
  }
  __syncthreads();
}

// ---------- fused rollout kernel: one wave per 16 batch rows ----------
__global__ __launch_bounds__(32)
void pusht_fused(const float* __restrict__ cls,
                 const float* __restrict__ aq, const float* __restrict__ av,
                 const float* __restrict__ bq0, const float* __restrict__ bv0,
                 const float* __restrict__ zb1, const float* __restrict__ zb2,
                 const float* __restrict__ sb1, const float* __restrict__ sb2,
                 const float* __restrict__ fb1, const float* __restrict__ fb2,
                 const float* __restrict__ fb3,
                 const float* __restrict__ kWv, const float* __restrict__ kbv,
                 const float* __restrict__ mWv, const float* __restrict__ mbv,
                 const float* __restrict__ ob1, const float* __restrict__ ob2,
                 const _Float16* __restrict__ wsH,
                 float* __restrict__ loP, float* __restrict__ lpP, float* __restrict__ lkP)
{
  __shared__ __align__(16) _Float16 aS[16*AS];   // input tile / observer x
  __shared__ __align__(16) _Float16 gS[16*GS];   // hidden activations (f16)
  __shared__ __align__(16) _Float16 fS[32*FS];   // f-net input (32 body-rows x 32)
  __shared__ __align__(16) float    cS[256];     // fp32 C-tile scratch

  const int lane = threadIdx.x;
  const int lm   = lane & 15;                    // batch row within tile
  const int bk   = lane >> 4;                    // body index (0=agent,1=block)
  const int hb   = bk << 3;                      // C-tile row base
  const int Rg   = blockIdx.x * 16;
  const int row  = Rg + lm;

  // stage first = cls_embs[:,0,:] tile as f16
  for (int idx = lane; idx < 16*EDIM; idx += 32){
    int r = idx / EDIM, c = idx % EDIM;
    aS[r*AS + c] = (_Float16)cls[((size_t)(Rg + r)*TSEQ)*EDIM + c];
  }
  __syncthreads();

  float z[8], sig[8];
  encoder16(aS, gS, cS, wsH+OFF_ZW1, wsH+OFF_ZW2, zb1, zb2, lane, false, z);
  encoder16(aS, gS, cS, wsH+OFF_SW1, wsH+OFF_SW2, sb1, sb2, lane, true,  sig);

  // z is constant over the rollout -> contact stiffness & mass once
  float kc, mm;
  { float s = kbv[0]; for (int i=0;i<8;++i) s += z[i]*kWv[i]; kc = softplus_f(s); }
  { float s = mbv[0]; for (int i=0;i<8;++i) s += z[i]*mWv[i]; mm = softplus_f(s); }

  // initial stacked state
  float qx,qy,vx,vy;
  if (bk == 0){
    size_t o = ((size_t)row*TSEQ)*2;
    qx=aq[o]; qy=aq[o+1]; vx=av[o]; vy=av[o+1];
  } else {
    qx=bq0[2*row]; qy=bq0[2*row+1]; vx=bv0[2*row]; vy=bv0[2*row+1];
  }
  float vpx=vx, vpy=vy, cprev=0.f;

  float e0 = 0.5f*mm*(vx*vx+vy*vy);
  float Eprev = e0 + __shfl_xor(e0, 16);

  float lo=0.f, lp=0.f, lk=0.f;

  // static parts of staging buffers: f-net z features + zero K-padding;
  // observer x K-padding (cols 8..31)
  for (int i=0;i<8;++i)  fS[lane*FS + 4 + i] = (_Float16)z[i];
  for (int c=12;c<32;++c) fS[lane*FS + c] = (_Float16)0.f;
  for (int c=8+bk*12; c<8+bk*12+12; ++c) aS[lm*AS + c] = (_Float16)0.f;
  __syncthreads();

  for (int step=0; step<NSTEP; ++step){
    // pairwise (K=2: single off-diagonal partner via cross-lane xor)
    float qox = __shfl_xor(qx,16), qoy = __shfl_xor(qy,16);
    float dx = qx-qox, dy = qy-qoy;
    float dist = sqrtf(dx*dx + dy*dy + EPS_C);
    float cont = sigmoid_f((CONTACT_C - dist)/TAU_C);
    float ax = (step==0)?0.f:(vx-vpx)/DT_C;
    float ay = (step==0)?0.f:(vy-vpy)/DT_C;
    float innov = fmaxf(cont - cprev, 0.f) * sqrtf(ax*ax + ay*ay + EPS_C);
    vpx=vx; vpy=vy;

    // ---- f-net: rows = 32 (16 batch rows x 2 bodies), 12->64->64->2 ----
    fS[lane*FS+0]=(_Float16)qx; fS[lane*FS+1]=(_Float16)qy;
    fS[lane*FS+2]=(_Float16)vx; fS[lane*FS+3]=(_Float16)vy;
    __syncthreads();
    float flx=0.f, fly=0.f;
#pragma unroll
    for (int mt=0; mt<2; ++mt){
#pragma unroll
      for (int nt=0; nt<4; ++nt){                         // h1 = gelu(x@fW1+b)
        v8f acc = splat8(fb1[nt*16+lm]);
        acc = wmma_f16(frag_a(fS + mt*16*FS, lm, FS, 0, 0, lane),
                       frag_b(wsH+OFF_FW1, 0, nt, 4, lane), acc);
        int cc = nt*16+lm;
#pragma unroll
        for (int r=0;r<8;++r) gS[(hb+r)*GS + cc] = (_Float16)gelu_t(acc[r]);
      }
      __syncthreads();
#pragma unroll
      for (int nt=0; nt<4; ++nt){                         // h2 = gelu(h1@fW2+b)
        v8f acc = splat8(fb2[nt*16+lm]);
#pragma unroll
        for (int kt=0; kt<2; ++kt)
          acc = wmma_f16(frag_a(gS, lm, GS, 0, kt*32, lane),
                         frag_b(wsH+OFF_FW2, kt, nt, 4, lane), acc);
        int cc = 128 + nt*16 + lm;
#pragma unroll
        for (int r=0;r<8;++r) gS[(hb+r)*GS + cc] = (_Float16)gelu_t(acc[r]);
      }
      __syncthreads();
      {                                                   // f = h2@fW3+b (2 cols)
        v8f acc = splat8((lm<2) ? fb3[lm] : 0.f);
#pragma unroll
        for (int kt=0; kt<2; ++kt)
          acc = wmma_f16(frag_a(gS, lm, GS, 128, kt*32, lane),
                         frag_b(wsH+OFF_FW3, kt, 0, 1, lane), acc);
#pragma unroll
        for (int r=0;r<8;++r) cS[(hb+r)*16 + lm] = acc[r];
      }
      __syncthreads();
      if (bk == mt){ flx = cS[lm*16+0]; fly = cS[lm*16+1]; }
      __syncthreads();
    }

    // ---- physics integrate + sigma gate ----
    float pen = fmaxf(CONTACT_C - dist, 0.f);
    float fs  = kc * cont * pen / dist;
    float vnx = vx + DT_C*(flx + fs*dx);
    float vny = vy + DT_C*(fly + fs*dy);
    float qnx = qx + DT_C*vnx;
    float qny = qy + DT_C*vny;
    float gate = sigmoid_f((innov - INNOV_C)*10.f);
    float sf = 1.f - 0.5f*gate;
#pragma unroll
    for (int i=0;i<8;++i) sig[i] *= sf;

    // restack: agent from sequence, block simulated
    if (bk == 0){
      size_t o = ((size_t)row*TSEQ + (step+1))*2;
      qx=aq[o]; qy=aq[o+1]; vx=av[o]; vy=av[o+1];
    } else { qx=qnx; qy=qny; vx=vnx; vy=vny; }
    cprev = cont;

    // ---- observer: x(16x8) -> gelu(256) -> pred(192), fused loss ----
    aS[lm*AS + bk*4 + 0]=(_Float16)qx; aS[lm*AS + bk*4 + 1]=(_Float16)qy;
    aS[lm*AS + bk*4 + 2]=(_Float16)vx; aS[lm*AS + bk*4 + 3]=(_Float16)vy;
    __syncthreads();
#pragma unroll
    for (int nt=0; nt<16; ++nt){
      v8f acc = splat8(ob1[nt*16+lm]);
      acc = wmma_f16(frag_a(aS, lm, AS, 0, 0, lane),
                     frag_b(wsH+OFF_OW1, 0, nt, 16, lane), acc);
      int cc = nt*16+lm;
#pragma unroll
      for (int r=0;r<8;++r) gS[(hb+r)*GS + cc] = (_Float16)gelu_t(acc[r]);
    }
    __syncthreads();
    for (int nt=0; nt<12; ++nt){
      int col = nt*16 + lm;
      v8f acc = splat8(ob2[col]);
#pragma unroll
      for (int kt=0; kt<8; ++kt)
        acc = wmma_f16(frag_a(gS, lm, GS, 0, kt*32, lane),
                       frag_b(wsH+OFF_OW2, kt, nt, 12, lane), acc);
#pragma unroll
      for (int r=0;r<8;++r){
        float t = cls[((size_t)(Rg + hb + r)*TSEQ + (step+1))*EDIM + col];
        float d = acc[r] - t;
        lo += d*d;
      }
    }
    __syncthreads();

    // ---- energy & passivity loss (row energy duplicated across lane pair) ----
    float el = 0.5f*mm*(vx*vx + vy*vy);
    float Er = el + __shfl_xor(el, 16);
    lp += fmaxf(Er - Eprev, 0.f);
    Eprev = Er;
  }

  // KL (final sigma, constant z); each lane owns distinct (row,body,8 dims)
#pragma unroll
  for (int i=0;i<8;++i){
    float s = sig[i];
    lk += s*s + z[i]*z[i] - 1.f - 2.f*__logf(s + EPS_C);
  }

  // wave reduction, per-block partials (deterministic; no float atomics)
#pragma unroll
  for (int off=16; off; off>>=1){
    lo += __shfl_xor(lo, off);
    lp += __shfl_xor(lp, off);
    lk += __shfl_xor(lk, off);
  }
  if (lane == 0){ loP[blockIdx.x]=lo; lpP[blockIdx.x]=lp; lkP[blockIdx.x]=lk; }
}

// ---------- deterministic final reduction ----------
__global__ void finalize_loss(const float* __restrict__ loP, const float* __restrict__ lpP,
                              const float* __restrict__ lkP, float* __restrict__ out){
  __shared__ float s0[256], s1[256], s2[256];
  int t = threadIdx.x;
  float a=0.f,b=0.f,c=0.f;
  for (int i=t; i<NTILES; i+=256){ a+=loP[i]; b+=lpP[i]; c+=lkP[i]; }
  s0[t]=a; s1[t]=b; s2[t]=c; __syncthreads();
  for (int off=128; off; off>>=1){
    if (t<off){ s0[t]+=s0[t+off]; s1[t]+=s1[t+off]; s2[t]+=s2[t+off]; }
    __syncthreads();
  }
  if (t==0){
    float lo = s0[0], lp = s1[0], lk = s2[0];
    float loss_obs  = lo / (32768.0f*4.0f*192.0f);
    float loss_pass = (0.5f*lp) / (4.0f*32768.0f);     // *0.5: row energy double-counted
    float loss_kl   = 0.5f*lk / (32768.0f*2.0f*8.0f);
    out[0] = loss_obs + 0.1f*loss_pass + 0.01f*loss_kl;
  }
}

extern "C" void kernel_launch(void* const* d_in, const int* in_sizes, int n_in,
                              void* d_out, int out_size, void* d_ws, size_t ws_size,
                              hipStream_t stream) {
  (void)in_sizes; (void)n_in; (void)out_size; (void)ws_size;
  const float* cls = (const float*)d_in[0];
  const float* aq  = (const float*)d_in[1];
  const float* av  = (const float*)d_in[2];
  const float* bq0 = (const float*)d_in[3];
  const float* bv0 = (const float*)d_in[4];
  const float* zW1 = (const float*)d_in[5];  const float* zb1 = (const float*)d_in[6];
  const float* zW2 = (const float*)d_in[7];  const float* zb2 = (const float*)d_in[8];
  const float* sW1 = (const float*)d_in[9];  const float* sb1 = (const float*)d_in[10];
  const float* sW2 = (const float*)d_in[11]; const float* sb2 = (const float*)d_in[12];
  const float* fW1 = (const float*)d_in[13]; const float* fb1 = (const float*)d_in[14];
  const float* fW2 = (const float*)d_in[15]; const float* fb2 = (const float*)d_in[16];
  const float* fW3 = (const float*)d_in[17]; const float* fb3 = (const float*)d_in[18];
  const float* kWv = (const float*)d_in[19]; const float* kbv = (const float*)d_in[20];
  const float* mWv = (const float*)d_in[21]; const float* mbv = (const float*)d_in[22];
  const float* oW1 = (const float*)d_in[23]; const float* ob1 = (const float*)d_in[24];
  const float* oW2 = (const float*)d_in[25]; const float* ob2 = (const float*)d_in[26];

  // workspace: [0,3*NTILES) fp32 loss partials, then packed f16 weights
  float* loP = (float*)d_ws;
  float* lpP = loP + NTILES;
  float* lkP = loP + 2*NTILES;
  _Float16* wsH = (_Float16*)((char*)d_ws + 3*NTILES*sizeof(float));

  struct PackDesc { const float* src; int K, N, KT, NT, off; };
  const PackDesc packs[9] = {
    { zW1, 192, 128, 6,  8, OFF_ZW1 },
    { zW2, 128,  16, 4,  1, OFF_ZW2 },
    { sW1, 192, 128, 6,  8, OFF_SW1 },
    { sW2, 128,  16, 4,  1, OFF_SW2 },
    { fW1,  12,  64, 1,  4, OFF_FW1 },
    { fW2,  64,  64, 2,  4, OFF_FW2 },
    { fW3,  64,   2, 2,  1, OFF_FW3 },
    { oW1,   8, 256, 1, 16, OFF_OW1 },
    { oW2, 256, 192, 8, 12, OFF_OW2 },
  };
  for (int i=0;i<9;++i){
    int total = packs[i].KT * packs[i].NT * 512;
    pack_wfrag<<<(total+255)/256, 256, 0, stream>>>(packs[i].src, wsH + packs[i].off,
                                                    packs[i].K, packs[i].N,
                                                    packs[i].NT, total);
  }

  pusht_fused<<<NTILES, 32, 0, stream>>>(cls, aq, av, bq0, bv0,
      zb1, zb2, sb1, sb2, fb1, fb2, fb3,
      kWv, kbv, mWv, mbv, ob1, ob2,
      wsH, loP, lpP, lkP);

  finalize_loss<<<1, 256, 0, stream>>>(loP, lpP, lkP, (float*)d_out);
}
